// MultiAttention_34368328302815
// MI455X (gfx1250) — compile-verified, matrix operand order checked by reading, not run on hardware
//
#include <hip/hip_runtime.h>

// ---------- CDNA5 WMMA types ----------
typedef __attribute__((ext_vector_type(16))) __bf16          v16bf;
typedef __attribute__((ext_vector_type(8)))  float           v8f;
typedef __attribute__((ext_vector_type(8)))  unsigned short  ush8;

union Frag32 { ush8 h[2]; v16bf v; };

#define WMMA_BF16(A, B, C) \
  __builtin_amdgcn_wmma_f32_16x16x32_bf16(false, (A), false, (B), (short)0, (C), false, false)

// ---------- gfx1250 async global->LDS staging (guarded) ----------
#if defined(__has_builtin)
#  if __has_builtin(__builtin_amdgcn_global_load_async_to_lds_b128)
#    define ASYNC_LDS 1
#  endif
#endif
#ifndef ASYNC_LDS
#  define ASYNC_LDS 0
#endif

// Builtin parameter type per hipcc diagnostic:
// 'int __attribute__((__vector_size__(4 * sizeof(int)))) *'
typedef __attribute__((__vector_size__(4 * sizeof(int)))) int vsi4;

// Copy one 16-byte chunk global -> LDS (per active lane).
__device__ __forceinline__ void stage16(const unsigned short* g, unsigned short* l) {
#if ASYNC_LDS
  __builtin_amdgcn_global_load_async_to_lds_b128((vsi4*)g, (vsi4*)l, 0, 0);
#else
  *(ush8*)l = *(const ush8*)g;
#endif
}

__device__ __forceinline__ void wait_stage() {
#if ASYNC_LDS
#  if __has_builtin(__builtin_amdgcn_s_wait_asynccnt)
  __builtin_amdgcn_s_wait_asynccnt(0);
#  else
  asm volatile("s_wait_asynccnt 0" ::: "memory");
#  endif
#endif
}

__device__ __forceinline__ unsigned short f2bf(float x) {
  unsigned u = __builtin_bit_cast(unsigned, x);
  u += 0x7FFFu + ((u >> 16) & 1u);  // round-to-nearest-even
  return (unsigned short)(u >> 16);
}

// Load one bf16 A/B fragment (16x32 per-lane layout): lane holds row `row`,
// two 16B chunks at K = k0 and K = k0+16 (k0 already includes +8 for lanes>=16).
__device__ __forceinline__ v16bf load_frag(const unsigned short* base,
                                           int stride, int row, int k0) {
  const unsigned short* p = base + (size_t)row * stride + k0;
  Frag32 f;
  f.h[0] = *(const ush8*)(p);
  f.h[1] = *(const ush8*)(p + 16);
  return f.v;
}

// ---------- fp32 -> bf16 cast ----------
__global__ void f32_to_bf16(const float* __restrict__ src,
                            unsigned short* __restrict__ dst, int n) {
  int i = blockIdx.x * blockDim.x + threadIdx.x;
  if (i < n) dst[i] = f2bf(src[i]);
}

// ---------- WMMA GEMM: out = A(bf16,[M,K]) * W(bf16,[N,K])^T ----------
// Block = 128x128, 8 waves (4M x 2N), wave tile 32x64.
// A and B 128x32 k-slices double-buffered in LDS via async-to-LDS.
// MODE 0: bf16 out -> [b,h,s,d]  (Q,K)   MODE 1: bf16 -> [b,h,d,s] (V^T)
// MODE 2: fp32 out row-major + bias      (output projection)
template <int MODE>
__global__ __launch_bounds__(256)
void gemm_bf16(const unsigned short* __restrict__ A,
               const unsigned short* __restrict__ W,
               void* __restrict__ outp,
               const float* __restrict__ bias,
               int M, int N, int K) {
  __shared__ __align__(16) unsigned short Alds[2][128 * 32];
  __shared__ __align__(16) unsigned short Blds[2][128 * 32];

  const int tid   = threadIdx.x;
  const int lane  = tid & 31;
  const int lhalf = lane >> 4;
  const int koff  = lhalf * 8;
  const int ln    = lane & 15;
  const int wave  = tid >> 5;
  const int wm    = wave & 3, wn = wave >> 2;

  const int mb0 = blockIdx.y * 128;
  const int nb0 = blockIdx.x * 128;

  // staging map: 512 chunks of 16B per 128x32 slice -> 2 chunks/thread
  const int r0 = (tid) >> 2,        kc0 = ((tid) & 3) * 8;
  const int r1 = (tid + 256) >> 2,  kc1 = ((tid + 256) & 3) * 8;

  const v8f vzero = {0.f, 0.f, 0.f, 0.f, 0.f, 0.f, 0.f, 0.f};
  v8f acc[2][4];
#pragma unroll
  for (int i = 0; i < 2; ++i)
#pragma unroll
    for (int j = 0; j < 4; ++j) acc[i][j] = vzero;

  // prologue: stage k-slice 0 into buffer 0
  stage16(A + (size_t)(mb0 + r0) * K + kc0, &Alds[0][r0 * 32 + kc0]);
  stage16(A + (size_t)(mb0 + r1) * K + kc1, &Alds[0][r1 * 32 + kc1]);
  stage16(W + (size_t)(nb0 + r0) * K + kc0, &Blds[0][r0 * 32 + kc0]);
  stage16(W + (size_t)(nb0 + r1) * K + kc1, &Blds[0][r1 * 32 + kc1]);

  for (int k0 = 0, it = 0; k0 < K; k0 += 32, ++it) {
    const int cur = it & 1;
    wait_stage();
    __syncthreads();  // buffer `cur` ready everywhere; prev compute finished
    if (k0 + 32 < K) {
      const int nxt = cur ^ 1, kn = k0 + 32;
      stage16(A + (size_t)(mb0 + r0) * K + kn + kc0, &Alds[nxt][r0 * 32 + kc0]);
      stage16(A + (size_t)(mb0 + r1) * K + kn + kc1, &Alds[nxt][r1 * 32 + kc1]);
      stage16(W + (size_t)(nb0 + r0) * K + kn + kc0, &Blds[nxt][r0 * 32 + kc0]);
      stage16(W + (size_t)(nb0 + r1) * K + kn + kc1, &Blds[nxt][r1 * 32 + kc1]);
    }
    v16bf af[2];
#pragma unroll
    for (int i = 0; i < 2; ++i)
      af[i] = load_frag(&Alds[cur][0], 32, wm * 32 + i * 16 + ln, koff);
    v16bf bfr[4];
#pragma unroll
    for (int j = 0; j < 4; ++j)
      bfr[j] = load_frag(&Blds[cur][0], 32, wn * 64 + j * 16 + ln, koff);
#pragma unroll
    for (int i = 0; i < 2; ++i)
#pragma unroll
      for (int j = 0; j < 4; ++j)
        acc[i][j] = WMMA_BF16(af[i], bfr[j], acc[i][j]);
  }

  // C/D layout: VGPR e -> row (e + 8*lhalf), column = lane&15.
#pragma unroll
  for (int i = 0; i < 2; ++i) {
#pragma unroll
    for (int j = 0; j < 4; ++j) {
#pragma unroll
      for (int e = 0; e < 8; ++e) {
        const int m = mb0 + wm * 32 + i * 16 + e + lhalf * 8;
        const int n = nb0 + wn * 64 + j * 16 + ln;
        const float val = acc[i][j][e];
        if constexpr (MODE == 2) {
          ((float*)outp)[(size_t)m * N + n] = val + bias[n];
        } else {
          const int b = m >> 11, s = m & 2047;  // m over B*S (S=2048)
          const int h = n >> 6,  d = n & 63;    // n over H*Dh (Dh=64)
          unsigned short* o16 = (unsigned short*)outp;
          if constexpr (MODE == 0)
            o16[(((size_t)(b * 16 + h) * 2048) + s) * 64 + d] = f2bf(val);
          else
            o16[(((size_t)(b * 16 + h) * 64) + d) * 2048 + s] = f2bf(val);
        }
      }
    }
  }
}

// ---------- causal flash attention ----------
// Q,K: [b,h,s,64] bf16. Vt: [b,h,64,s] bf16. attn out: [b,s,1024] bf16.
// Block = 8 waves x 16 query rows = 128 rows per (b,h). K/V tiles staged in
// LDS and shared by all waves; block-uniform j loop, element masks handle
// the causal diagonal (fully-masked tiles degenerate to corr=1, p=0).
__global__ __launch_bounds__(256)
void flash_attn(const unsigned short* __restrict__ Q,
                const unsigned short* __restrict__ Kh,
                const unsigned short* __restrict__ Vt,
                unsigned short* __restrict__ attn) {
  __shared__ __align__(16) unsigned short Klds[32 * 64];      // keys x d
  __shared__ __align__(16) unsigned short Vlds[64 * 32];      // d x keys
  __shared__ __align__(16) unsigned short Plds[8 * 16 * 32];  // per-wave P

  const int tid   = threadIdx.x;
  const int lane  = tid & 31;
  const int lhalf = lane >> 4;
  const int koff  = lhalf * 8;
  const int ln    = lane & 15;
  const int wave  = tid >> 5;

  const int bh    = blockIdx.y;
  const int q0    = blockIdx.x * 128;
  const int qbase = q0 + wave * 16;

  const unsigned short* Qp = Q  + (size_t)bh * 2048 * 64;
  const unsigned short* Kp = Kh + (size_t)bh * 2048 * 64;
  const unsigned short* Vp = Vt + (size_t)bh * 64 * 2048;

  v16bf qf[2];
  qf[0] = load_frag(Qp, 64, qbase + ln, koff);
  qf[1] = load_frag(Qp, 64, qbase + ln, 32 + koff);

  const v8f vzero = {0.f, 0.f, 0.f, 0.f, 0.f, 0.f, 0.f, 0.f};
  float mrow[8], lrow[8];
#pragma unroll
  for (int e = 0; e < 8; ++e) { mrow[e] = -1e30f; lrow[e] = 0.0f; }
  v8f o[4];
#pragma unroll
  for (int d = 0; d < 4; ++d) o[d] = vzero;

  unsigned short* P = &Plds[wave * 512];
  const float scale = 0.125f;  // 1/sqrt(64)

  // staging maps: K tile 32x64 -> 256 chunks; V tile 64x32 -> 256 chunks
  const int krow = tid >> 3, kkc = (tid & 7) * 8;
  const int vrow = tid >> 2, vkc = (tid & 3) * 8;

  const int jend = q0 + 128;  // block-uniform
  for (int j0 = 0; j0 < jend; j0 += 32) {
    stage16(Kp + (size_t)(j0 + krow) * 64 + kkc, &Klds[krow * 64 + kkc]);
    stage16(Vp + (size_t)vrow * 2048 + j0 + vkc, &Vlds[vrow * 32 + vkc]);
    wait_stage();
    __syncthreads();  // tiles ready

    // S tile 16x32 = Q(16x64) * K(32x64)^T, K fragments from LDS
    v8f s0 = vzero, s1 = vzero;
    v16bf kf;
    kf = load_frag(&Klds[0], 64, ln,      koff);      s0 = WMMA_BF16(qf[0], kf, s0);
    kf = load_frag(&Klds[0], 64, ln,      32 + koff); s0 = WMMA_BF16(qf[1], kf, s0);
    kf = load_frag(&Klds[0], 64, 16 + ln, koff);      s1 = WMMA_BF16(qf[0], kf, s1);
    kf = load_frag(&Klds[0], 64, 16 + ln, 32 + koff); s1 = WMMA_BF16(qf[1], kf, s1);

    // online softmax; row reductions across the 16 lanes of this half
#pragma unroll
    for (int e = 0; e < 8; ++e) {
      const int m = qbase + e + lhalf * 8;
      float v0 = s0[e] * scale;
      float v1 = s1[e] * scale;
      if (j0 + ln      > m) v0 = -1e30f;
      if (j0 + 16 + ln > m) v1 = -1e30f;
      float cm = fmaxf(v0, v1);
#pragma unroll
      for (int off = 1; off < 16; off <<= 1) cm = fmaxf(cm, __shfl_xor(cm, off, 32));
      const float mnew = fmaxf(mrow[e], cm);
      const float corr = __expf(mrow[e] - mnew);
      const float p0 = __expf(v0 - mnew);
      const float p1 = __expf(v1 - mnew);
      float rs = p0 + p1;
#pragma unroll
      for (int off = 1; off < 16; off <<= 1) rs += __shfl_xor(rs, off, 32);
      lrow[e] = lrow[e] * corr + rs;
      mrow[e] = mnew;
#pragma unroll
      for (int d = 0; d < 4; ++d) o[d][e] *= corr;
      P[(e + lhalf * 8) * 32 + ln]      = f2bf(p0);
      P[(e + lhalf * 8) * 32 + 16 + ln] = f2bf(p1);
    }

    // reread P as A-fragment; O += P * V with V fragments from LDS
    Frag32 pf;
    pf.h[0] = *(const ush8*)&P[ln * 32 + koff];
    pf.h[1] = *(const ush8*)&P[ln * 32 + koff + 16];
#pragma unroll
    for (int d = 0; d < 4; ++d) {
      v16bf vf = load_frag(&Vlds[0], 32, d * 16 + ln, koff);
      o[d] = WMMA_BF16(pf.v, vf, o[d]);
    }

    __syncthreads();  // everyone done reading K/V before next overwrite
  }

  const int b = bh >> 4, h = bh & 15;
#pragma unroll
  for (int d = 0; d < 4; ++d) {
#pragma unroll
    for (int e = 0; e < 8; ++e) {
      const int m = qbase + e + lhalf * 8;
      const float val = o[d][e] / lrow[e];
      attn[((size_t)b * 2048 + m) * 1024 + h * 64 + d * 16 + ln] = f2bf(val);
    }
  }
}

// ---------- launcher ----------
extern "C" void kernel_launch(void* const* d_in, const int* in_sizes, int n_in,
                              void* d_out, int out_size, void* d_ws, size_t ws_size,
                              hipStream_t stream) {
  (void)in_sizes; (void)n_in; (void)out_size; (void)ws_size;
  const float* x  = (const float*)d_in[0];
  const float* Wq = (const float*)d_in[1];
  const float* Wk = (const float*)d_in[2];
  const float* Wv = (const float*)d_in[3];
  const float* Wo = (const float*)d_in[4];
  const float* bo = (const float*)d_in[5];

  const int M = 4096, N = 1024, K = 1024;  // M = B*S
  char* ws = (char*)d_ws;
  const size_t MB = 1024 * 1024;
  unsigned short* Xb  = (unsigned short*)(ws + 0 * MB);   // 8 MB
  unsigned short* Wqb = (unsigned short*)(ws + 8 * MB);   // 2 MB
  unsigned short* Wkb = (unsigned short*)(ws + 10 * MB);
  unsigned short* Wvb = (unsigned short*)(ws + 12 * MB);
  unsigned short* Wob = (unsigned short*)(ws + 14 * MB);
  unsigned short* Qh  = (unsigned short*)(ws + 16 * MB);  // [b,h,s,d] 8 MB
  unsigned short* Kc  = (unsigned short*)(ws + 24 * MB);  // [b,h,s,d] 8 MB
  unsigned short* Vt  = (unsigned short*)(ws + 32 * MB);  // [b,h,d,s] 8 MB
  unsigned short* At  = (unsigned short*)(ws + 40 * MB);  // [b,s,D]   8 MB

  f32_to_bf16<<<(M * K + 255) / 256, 256, 0, stream>>>(x,  Xb,  M * K);
  f32_to_bf16<<<(N * K + 255) / 256, 256, 0, stream>>>(Wq, Wqb, N * K);
  f32_to_bf16<<<(N * K + 255) / 256, 256, 0, stream>>>(Wk, Wkb, N * K);
  f32_to_bf16<<<(N * K + 255) / 256, 256, 0, stream>>>(Wv, Wvb, N * K);
  f32_to_bf16<<<(N * K + 255) / 256, 256, 0, stream>>>(Wo, Wob, N * K);

  dim3 grid(N / 128, M / 128), blk(256);
  gemm_bf16<0><<<grid, blk, 0, stream>>>(Xb, Wqb, Qh, nullptr, M, N, K);
  gemm_bf16<0><<<grid, blk, 0, stream>>>(Xb, Wkb, Kc, nullptr, M, N, K);
  gemm_bf16<1><<<grid, blk, 0, stream>>>(Xb, Wvb, Vt, nullptr, M, N, K);

  flash_attn<<<dim3(16, 32), blk, 0, stream>>>(Qh, Kc, Vt, At);

  gemm_bf16<2><<<grid, blk, 0, stream>>>(At, Wob, d_out, bo, M, N, K);
}
